// CausalWanAttentionBlock_8615704395908
// MI455X (gfx1250) — matrix operand order repaired and checked by reading, not compiled, not run on hardware
//
#include <hip/hip_runtime.h>
#include <hip/hip_bf16.h>

// ---------------------------------------------------------------------------
// Problem constants (CausalWanAttentionBlock)
// ---------------------------------------------------------------------------
#define DIM   1536
#define HEADS 12
#define HD    128
#define FFN_D 8960
#define NF    6
#define GH    20
#define GW    44
#define S_TOK (NF * GH * GW)   // 5280
#define FS    (GH * GW)        // 880 tokens per frame
#define LCTX  512
#define EPS   1e-6f

typedef _Float16 v16h __attribute__((ext_vector_type(16)));
typedef _Float16 v8h  __attribute__((ext_vector_type(8)));
typedef float    v8f  __attribute__((ext_vector_type(8)));
typedef unsigned int u32x4 __attribute__((ext_vector_type(4)));
typedef unsigned int u32x8 __attribute__((ext_vector_type(8)));

__device__ __forceinline__ v16h load_frag(const _Float16* p0, const _Float16* p1) {
    v8h lo = *(const v8h*)p0;
    v8h hi = *(const v8h*)p1;
    return __builtin_shufflevector(lo, hi, 0,1,2,3,4,5,6,7,8,9,10,11,12,13,14,15);
}

__device__ __forceinline__ v8f wmma_f16(v16h a, v16h b, v8f c) {
    return __builtin_amdgcn_wmma_f32_16x16x32_f16(false, a, false, b, (short)0, c, false, false);
}

// ---------------------------------------------------------------------------
// RoPE angle table: cos/sin of theta[s][j], j<64
// ---------------------------------------------------------------------------
__global__ void theta_kernel(const float* __restrict__ angle,
                             float* __restrict__ cosb, float* __restrict__ sinb) {
    int idx = blockIdx.x * blockDim.x + threadIdx.x;
    if (idx >= S_TOK * 64) return;
    int s = idx >> 6, j = idx & 63;
    int f = s / FS, rem = s % FS;
    int hh = rem / GW, ww = rem % GW;
    int row = (j < 22) ? f : (j < 43) ? hh : ww;   // c1=22, c2=21
    float th = angle[row * 64 + j];
    cosb[idx] = cosf(th);
    sinb[idx] = sinf(th);
}

// ---------------------------------------------------------------------------
// LayerNorm + (1+scale)*x + shift  (scale/shift = modulation + e, per frame)
// ---------------------------------------------------------------------------
__global__ __launch_bounds__(256)
void ln_mod(const float* __restrict__ X, const float* __restrict__ e,
            const float* __restrict__ modu, int ishift, int iscale,
            _Float16* __restrict__ out) {
    __shared__ float s_sum[8], s_sq[8];
    int s = blockIdx.x;
    int tid = threadIdx.x;
    const float* xr = X + (size_t)s * DIM;
    float sum = 0.f, sq = 0.f;
    float vals[6];
#pragma unroll
    for (int i = 0; i < 6; ++i) {
        float v = xr[tid + i * 256];
        vals[i] = v; sum += v; sq += v * v;
    }
#pragma unroll
    for (int off = 1; off < 32; off <<= 1) {
        sum += __shfl_xor(sum, off, 32);
        sq  += __shfl_xor(sq,  off, 32);
    }
    if ((tid & 31) == 0) { s_sum[tid >> 5] = sum; s_sq[tid >> 5] = sq; }
    __syncthreads();
    float ts = 0.f, tq = 0.f;
#pragma unroll
    for (int i = 0; i < 8; ++i) { ts += s_sum[i]; tq += s_sq[i]; }
    float mean = ts * (1.0f / DIM);
    float var  = tq * (1.0f / DIM) - mean * mean;
    float rinv = rsqrtf(var + EPS);
    int frame = s / FS;
    const float* eb = e + (size_t)frame * 6 * DIM;
    _Float16* orow = out + (size_t)s * DIM;
#pragma unroll
    for (int i = 0; i < 6; ++i) {
        int c = tid + i * 256;
        float sc = modu[iscale * DIM + c] + eb[iscale * DIM + c];
        float sh = modu[ishift * DIM + c] + eb[ishift * DIM + c];
        orow[c] = (_Float16)(((vals[i] - mean) * rinv) * (1.0f + sc) + sh);
    }
}

// ---------------------------------------------------------------------------
// WMMA GEMM: out = epi(A[f16, MxK] * W[f32->f16, KxN] + bias)
// block = 256 thr (8 waves), tile 128x64, BK=32, waves 4(M)x2(N) of 32x32
// A tile staged into LDS by the Tensor Data Mover (one descriptor per K-step,
// issued by wave 0; TDM pads each 64B row with 16B -> 80B LDS pitch, and
// zero-fills rows past the M tail via tensor_dim1).
// ---------------------------------------------------------------------------
#define BM 128
#define BN 64
#define BK 32
#define ASTR 40
#define BSTR 40

enum { EPI_F32 = 0, EPI_RES_GATE = 1, EPI_GELU_F16 = 2 };

template<int EPI>
__global__ __launch_bounds__(256)
void gemm_wmma(const _Float16* __restrict__ A, const float* __restrict__ W,
               const float* __restrict__ bias, void* __restrict__ outv,
               const float* __restrict__ res, const float* __restrict__ gmod,
               const float* __restrict__ ge, int M, int N, int K, int fs) {
    __shared__ _Float16 ash[BM * ASTR];
    __shared__ _Float16 bsh[BN * BSTR];
    int tid  = threadIdx.x;
    int lane = tid & 31, wave = tid >> 5;
    int bm0 = blockIdx.y * BM;
    int bn0 = blockIdx.x * BN;
    int wm = (wave & 3) * 32, wn = (wave >> 2) * 32;
    int l15 = lane & 15, lh = lane >> 4;
    v8f acc[2][2] = {};
    int br = tid >> 3;          // 0..31 (k row)
    int bc = (tid & 7) * 8;     // 0..56 (n col)

    unsigned lds_base = (unsigned)(size_t)(&ash[0]);
    unsigned rowbytes = (unsigned)K * 2u;
    unsigned remrows  = (unsigned)(M - bm0);

    for (int k0 = 0; k0 < K; k0 += BK) {
        // ---- A tile via Tensor Data Mover (wave 0 only; EXEC ignored by TDM)
        if (wave == 0) {
            unsigned long long ga =
                (unsigned long long)(size_t)(A + (size_t)bm0 * K + k0);
            u32x4 g0;
            g0[0] = 1u;                                   // count=1 valid descriptor
            g0[1] = lds_base;                             // lds_addr
            g0[2] = (unsigned)(ga & 0xffffffffu);         // global_addr[31:0]
            g0[3] = (unsigned)((ga >> 32) & 0x01ffffffu)  // global_addr[56:32]
                    | (2u << 30);                         // type = 2 (image)
            u32x8 g1;
            g1[0] = (1u << 20)        // pad_enable
                  | (3u << 22)        // pad_interval: 16 DWORDs (=64B row)
                  | (3u << 25);       // pad_amount: 4 DWORDs (=16B) -> 80B pitch
                                      // data_size=0 (1 byte), wg_mask=0
            g1[1] = (rowbytes & 0xffffu) << 16;                    // tensor_dim0[15:0]
            g1[2] = (rowbytes >> 16) | ((remrows & 0xffffu) << 16); // dim0[31:16], dim1[15:0]
            g1[3] = (remrows >> 16) | (64u << 16);                  // dim1[31:16], tile_dim0=64B
            g1[4] = 128u;                                           // tile_dim1=128 rows
            g1[5] = rowbytes;                                       // dim0_stride[31:0]
            g1[6] = 0u;                                             // stride hi / dim1_stride lo
            g1[7] = 0u;
            asm volatile("tensor_load_to_lds %0, %1" :: "s"(g0), "s"(g1) : "memory");
        }
        // ---- B tile: transpose + fp32->f16 on VALU (weights are L2-resident)
        {
            const float4* p = (const float4*)(W + (size_t)(k0 + br) * N + bn0 + bc);
            float4 w0 = p[0], w1 = p[1];
            _Float16 h[8] = {(_Float16)w0.x,(_Float16)w0.y,(_Float16)w0.z,(_Float16)w0.w,
                             (_Float16)w1.x,(_Float16)w1.y,(_Float16)w1.z,(_Float16)w1.w};
#pragma unroll
            for (int j = 0; j < 8; ++j)
                bsh[(bc + j) * BSTR + br] = h[j];
            if (k0 + BK < K)
                __builtin_prefetch(W + (size_t)(k0 + BK + br) * N + bn0 + bc, 0, 0);
        }
        if (wave == 0) __builtin_amdgcn_s_wait_tensorcnt(0);
        __syncthreads();

        int kb = lh * 8;
        v16h afrag[2], bfrag[2];
#pragma unroll
        for (int si = 0; si < 2; ++si) {
            const _Float16* ap = &ash[(wm + si * 16 + l15) * ASTR];
            afrag[si] = load_frag(ap + kb, ap + 16 + kb);
        }
#pragma unroll
        for (int sj = 0; sj < 2; ++sj) {
            const _Float16* bp = &bsh[(wn + sj * 16 + l15) * BSTR + lh * 8];
            bfrag[sj] = load_frag(bp, bp + 16);
        }
#pragma unroll
        for (int si = 0; si < 2; ++si)
#pragma unroll
            for (int sj = 0; sj < 2; ++sj)
                acc[si][sj] = wmma_f16(afrag[si], bfrag[sj], acc[si][sj]);
        __syncthreads();
    }
    // epilogue
#pragma unroll
    for (int si = 0; si < 2; ++si) {
#pragma unroll
        for (int sj = 0; sj < 2; ++sj) {
            int n = bn0 + wn + sj * 16 + l15;
            float b = bias[n];
#pragma unroll
            for (int r = 0; r < 8; ++r) {
                int m = bm0 + wm + si * 16 + r + 8 * lh;
                if (m >= M) continue;
                float v = acc[si][sj][r] + b;
                size_t idx = (size_t)m * N + n;
                if (EPI == EPI_F32) {
                    ((float*)outv)[idx] = v;
                } else if (EPI == EPI_RES_GATE) {
                    float g = 1.0f;
                    if (gmod) g = gmod[n] + ge[(size_t)(m / fs) * (6 * DIM) + n];
                    ((float*)outv)[idx] = res[idx] + g * v;
                } else { // EPI_GELU_F16 (tanh approx)
                    float u = v + 0.044715f * v * v * v;
                    float t = 0.5f * v * (1.0f + tanhf(0.7978845608028654f * u));
                    ((_Float16*)outv)[idx] = (_Float16)t;
                }
            }
        }
    }
}

// ---------------------------------------------------------------------------
// RMS-norm (* gain) + optional RoPE. One 32-lane group per (token, head).
// out layout: [head][token][128] f16
// ---------------------------------------------------------------------------
__global__ __launch_bounds__(256)
void rms_rope(const float* __restrict__ X, const float* __restrict__ g,
              const float* __restrict__ cosb, const float* __restrict__ sinb,
              _Float16* __restrict__ out, int S, int rope) {
    int lane = threadIdx.x & 31, wid = threadIdx.x >> 5;
    int gid = blockIdx.x * 8 + wid;
    if (gid >= S * HEADS) return;
    int s = gid / HEADS, h = gid % HEADS;
    int d0 = lane * 4;
    const float* xp = X + (size_t)s * DIM + h * HD + d0;
    float v0 = xp[0], v1 = xp[1], v2 = xp[2], v3 = xp[3];
    float ss = v0*v0 + v1*v1 + v2*v2 + v3*v3;
#pragma unroll
    for (int off = 1; off < 32; off <<= 1) ss += __shfl_xor(ss, off, 32);
    float r = rsqrtf(ss * (1.0f / HD) + EPS);
    const float* gp = g + h * HD + d0;
    v0 *= r * gp[0]; v1 *= r * gp[1]; v2 *= r * gp[2]; v3 *= r * gp[3];
    if (rope) {
        int j0 = d0 >> 1;   // rope pair index = 2*lane
        float c0 = cosb[(size_t)s * 64 + j0],     sn0 = sinb[(size_t)s * 64 + j0];
        float c1 = cosb[(size_t)s * 64 + j0 + 1], sn1 = sinb[(size_t)s * 64 + j0 + 1];
        float a = v0, b = v1;
        v0 = a * c0 - b * sn0; v1 = a * sn0 + b * c0;
        a = v2; b = v3;
        v2 = a * c1 - b * sn1; v3 = a * sn1 + b * c1;
    }
    _Float16* op = out + ((size_t)h * S + s) * HD + d0;
    op[0] = (_Float16)v0; op[1] = (_Float16)v1;
    op[2] = (_Float16)v2; op[3] = (_Float16)v3;
}

// V pack: f32 [s][h*128+d] -> f16 Vt[h][d][s]
__global__ void vpack(const float* __restrict__ V, _Float16* __restrict__ Vt, int S) {
    int idx = blockIdx.x * blockDim.x + threadIdx.x;
    if (idx >= S * DIM) return;
    int s = idx / DIM, c = idx % DIM;
    int h = c >> 7, d = c & 127;
    Vt[((size_t)h * HD + d) * S + s] = (_Float16)V[idx];
}

__global__ void convert_f16(const float* __restrict__ in, _Float16* __restrict__ out, int n) {
    int idx = blockIdx.x * blockDim.x + threadIdx.x;
    if (idx < n) out[idx] = (_Float16)in[idx];
}

// ---------------------------------------------------------------------------
// Flash attention, one wave per 16 query rows, online softmax, frame-causal
// Q/K: [h][s][128] f16 ; Vt: [h][128][Svt] f16 ; out16: [s][h*128+d] f16
// ---------------------------------------------------------------------------
__global__ __launch_bounds__(256)
void flash_attn(const _Float16* __restrict__ Q, const _Float16* __restrict__ Km,
                const _Float16* __restrict__ Vt, _Float16* __restrict__ out16,
                int Sq, int Skv, int Svt, int fs /* 0 = dense */) {
    __shared__ _Float16 pbuf[8][16 * 40];
    int lane = threadIdx.x & 31, wave = threadIdx.x >> 5;
    int head = blockIdx.y;
    int qt = blockIdx.x * 8 + wave;
    if (qt * 16 >= Sq) return;
    int qbase = qt * 16;
    int l15 = lane & 15, lh = lane >> 4;

    v16h qf[4];
    {
        const _Float16* qp = Q + ((size_t)head * Sq + qbase + l15) * HD + 8 * lh;
#pragma unroll
        for (int f = 0; f < 4; ++f)
            qf[f] = load_frag(qp + 32 * f, qp + 32 * f + 16);
    }
    v8f o[8] = {};
    float rm[8], rl[8];
#pragma unroll
    for (int r = 0; r < 8; ++r) { rm[r] = -3.0e38f; rl[r] = 0.f; }

    int limit = (fs > 0) ? ((qbase / fs) + 1) * fs : Skv;
    int nb = (limit + 31) / 32;
    const float scale = 0.08838834764831845f;  // 1/sqrt(128)

    for (int b = 0; b < nb; ++b) {
        int jb = b * 32;
        v8f s0 = {}, s1 = {};
        const _Float16* kp0 = Km + ((size_t)head * Skv + jb + l15) * HD + 8 * lh;
        const _Float16* kp1 = kp0 + (size_t)16 * HD;
#pragma unroll
        for (int f = 0; f < 4; ++f) {
            v16h bk0 = load_frag(kp0 + 32 * f, kp0 + 32 * f + 16);
            s0 = wmma_f16(qf[f], bk0, s0);
            v16h bk1 = load_frag(kp1 + 32 * f, kp1 + 32 * f + 16);
            s1 = wmma_f16(qf[f], bk1, s1);
        }
        int j0 = jb + l15, j1 = jb + 16 + l15;
        float pv0[8], pv1[8];
#pragma unroll
        for (int r = 0; r < 8; ++r) {
            float a0 = s0[r] * scale; if (j0 >= limit) a0 = -3.0e38f;
            float a1 = s1[r] * scale; if (j1 >= limit) a1 = -3.0e38f;
            float t = fmaxf(a0, a1);
            t = fmaxf(t, __shfl_xor(t, 1, 32));
            t = fmaxf(t, __shfl_xor(t, 2, 32));
            t = fmaxf(t, __shfl_xor(t, 4, 32));
            t = fmaxf(t, __shfl_xor(t, 8, 32));
            float m2 = fmaxf(rm[r], t);
            float alpha = __expf(rm[r] - m2);
#pragma unroll
            for (int dt = 0; dt < 8; ++dt) o[dt][r] *= alpha;
            float p0 = __expf(a0 - m2);
            float p1 = __expf(a1 - m2);
            float su = p0 + p1;
            su += __shfl_xor(su, 1, 32);
            su += __shfl_xor(su, 2, 32);
            su += __shfl_xor(su, 4, 32);
            su += __shfl_xor(su, 8, 32);
            rl[r] = rl[r] * alpha + su;
            rm[r] = m2;
            pv0[r] = p0; pv1[r] = p1;
        }
        // repack P (16x32) through per-wave LDS into A-fragment layout
#pragma unroll
        for (int r = 0; r < 8; ++r) {
            int row = r + 8 * lh;
            pbuf[wave][row * 40 + l15]      = (_Float16)pv0[r];
            pbuf[wave][row * 40 + 16 + l15] = (_Float16)pv1[r];
        }
        asm volatile("s_wait_dscnt 0" ::: "memory");
        v16h pa;
        {
            const _Float16* pp = &pbuf[wave][l15 * 40 + 8 * lh];
            pa = load_frag(pp, pp + 16);
        }
#pragma unroll
        for (int dt = 0; dt < 8; ++dt) {
            const _Float16* vp = Vt + ((size_t)head * HD + dt * 16 + l15) * Svt + jb + 8 * lh;
            v16h bv = load_frag(vp, vp + 16);
            o[dt] = wmma_f16(pa, bv, o[dt]);
        }
    }
    // finalize: divide by row-sum, emit f16 [s][h*128+d]
#pragma unroll
    for (int r = 0; r < 8; ++r) {
        float inv = 1.0f / rl[r];
        int m = qbase + r + 8 * lh;
#pragma unroll
        for (int dt = 0; dt < 8; ++dt) {
            int d = dt * 16 + l15;
            out16[(size_t)m * DIM + head * HD + d] = (_Float16)(o[dt][r] * inv);
        }
    }
}

// ---------------------------------------------------------------------------
// Host-side orchestration
// ---------------------------------------------------------------------------
extern "C" void kernel_launch(void* const* d_in, const int* in_sizes, int n_in,
                              void* d_out, int out_size, void* d_ws, size_t ws_size,
                              hipStream_t stream) {
    (void)in_sizes; (void)n_in; (void)out_size; (void)ws_size;
    const float* x     = (const float*)d_in[0];
    const float* e     = (const float*)d_in[1];
    const float* ctx   = (const float*)d_in[2];
    const float* ang   = (const float*)d_in[3];
    const float* modu  = (const float*)d_in[7];
    const float* sa_wq = (const float*)d_in[8];
    const float* sa_bq = (const float*)d_in[9];
    const float* sa_wk = (const float*)d_in[10];
    const float* sa_bk = (const float*)d_in[11];
    const float* sa_wv = (const float*)d_in[12];
    const float* sa_bv = (const float*)d_in[13];
    const float* sa_wo = (const float*)d_in[14];
    const float* sa_bo = (const float*)d_in[15];
    const float* sa_gq = (const float*)d_in[16];
    const float* sa_gk = (const float*)d_in[17];
    const float* ca_wq = (const float*)d_in[18];
    const float* ca_bq = (const float*)d_in[19];
    const float* ca_wk = (const float*)d_in[20];
    const float* ca_bk = (const float*)d_in[21];
    const float* ca_wv = (const float*)d_in[22];
    const float* ca_bv = (const float*)d_in[23];
    const float* ca_wo = (const float*)d_in[24];
    const float* ca_bo = (const float*)d_in[25];
    const float* ca_gq = (const float*)d_in[26];
    const float* ca_gk = (const float*)d_in[27];
    const float* ffn_w1 = (const float*)d_in[28];
    const float* ffn_b1 = (const float*)d_in[29];
    const float* ffn_w2 = (const float*)d_in[30];
    const float* ffn_b2 = (const float*)d_in[31];
    float* xout = (float*)d_out;

    // workspace carve-out
    char* ws = (char*)d_ws;
    size_t off = 0;
    auto alloc = [&](size_t bytes) -> void* {
        void* p = ws + off;
        off += (bytes + 255) & ~(size_t)255;
        return p;
    };
    float*    cosb   = (float*)alloc((size_t)S_TOK * 64 * 4);
    float*    sinb   = (float*)alloc((size_t)S_TOK * 64 * 4);
    _Float16* y16    = (_Float16*)alloc((size_t)S_TOK * DIM * 2);
    float*    q32    = (float*)alloc((size_t)S_TOK * DIM * 4);
    float*    k32    = (float*)alloc((size_t)S_TOK * DIM * 4);
    float*    v32    = (float*)alloc((size_t)S_TOK * DIM * 4);
    _Float16* qh     = (_Float16*)alloc((size_t)S_TOK * DIM * 2);
    _Float16* kh     = (_Float16*)alloc((size_t)S_TOK * DIM * 2);
    _Float16* vt     = (_Float16*)alloc((size_t)S_TOK * DIM * 2);
    _Float16* attn16 = (_Float16*)alloc((size_t)S_TOK * DIM * 2);
    _Float16* x16    = (_Float16*)alloc((size_t)S_TOK * DIM * 2);
    _Float16* ctx16  = (_Float16*)alloc((size_t)LCTX * DIM * 2);
    _Float16* g16    = (_Float16*)alloc((size_t)S_TOK * FFN_D * 2);

    dim3 blk(256);
    dim3 gq(DIM / BN, (S_TOK + BM - 1) / BM);        // 24 x 42
    dim3 gc(DIM / BN, (LCTX + BM - 1) / BM);         // 24 x 4
    dim3 gf1(FFN_D / BN, (S_TOK + BM - 1) / BM);     // 140 x 42
    dim3 gfl((S_TOK / 16 + 7) / 8, HEADS);           // 42 x 12

    // ---- RoPE table
    theta_kernel<<<(S_TOK * 64 + 255) / 256, blk, 0, stream>>>(ang, cosb, sinb);
    // ---- y_in = mod(LN(x), e1, e0)
    ln_mod<<<S_TOK, blk, 0, stream>>>(x, e, modu, 0, 1, y16);
    // ---- self-attention QKV
    gemm_wmma<EPI_F32><<<gq, blk, 0, stream>>>(y16, sa_wq, sa_bq, q32, nullptr, nullptr, nullptr, S_TOK, DIM, DIM, FS);
    gemm_wmma<EPI_F32><<<gq, blk, 0, stream>>>(y16, sa_wk, sa_bk, k32, nullptr, nullptr, nullptr, S_TOK, DIM, DIM, FS);
    gemm_wmma<EPI_F32><<<gq, blk, 0, stream>>>(y16, sa_wv, sa_bv, v32, nullptr, nullptr, nullptr, S_TOK, DIM, DIM, FS);
    rms_rope<<<(S_TOK * HEADS + 7) / 8, blk, 0, stream>>>(q32, sa_gq, cosb, sinb, qh, S_TOK, 1);
    rms_rope<<<(S_TOK * HEADS + 7) / 8, blk, 0, stream>>>(k32, sa_gk, cosb, sinb, kh, S_TOK, 1);
    vpack<<<((size_t)S_TOK * DIM + 255) / 256, blk, 0, stream>>>(v32, vt, S_TOK);
    flash_attn<<<gfl, blk, 0, stream>>>(qh, kh, vt, attn16, S_TOK, S_TOK, S_TOK, FS);
    // ---- x = x + gate(attn @ Wo + bo, e2)
    gemm_wmma<EPI_RES_GATE><<<gq, blk, 0, stream>>>(attn16, sa_wo, sa_bo, xout, x,
                                                    modu + 2 * DIM, e + 2 * DIM, S_TOK, DIM, DIM, FS);
    // ---- cross-attention
    convert_f16<<<((size_t)S_TOK * DIM + 255) / 256, blk, 0, stream>>>(xout, x16, S_TOK * DIM);
    gemm_wmma<EPI_F32><<<gq, blk, 0, stream>>>(x16, ca_wq, ca_bq, q32, nullptr, nullptr, nullptr, S_TOK, DIM, DIM, FS);
    rms_rope<<<(S_TOK * HEADS + 7) / 8, blk, 0, stream>>>(q32, ca_gq, nullptr, nullptr, qh, S_TOK, 0);
    convert_f16<<<((size_t)LCTX * DIM + 255) / 256, blk, 0, stream>>>(ctx, ctx16, LCTX * DIM);
    gemm_wmma<EPI_F32><<<gc, blk, 0, stream>>>(ctx16, ca_wk, ca_bk, k32, nullptr, nullptr, nullptr, LCTX, DIM, DIM, FS);
    rms_rope<<<(LCTX * HEADS + 7) / 8, blk, 0, stream>>>(k32, ca_gk, nullptr, nullptr, kh, LCTX, 0);
    gemm_wmma<EPI_F32><<<gc, blk, 0, stream>>>(ctx16, ca_wv, ca_bv, v32, nullptr, nullptr, nullptr, LCTX, DIM, DIM, FS);
    vpack<<<((size_t)LCTX * DIM + 255) / 256, blk, 0, stream>>>(v32, vt, LCTX);
    flash_attn<<<gfl, blk, 0, stream>>>(qh, kh, vt, attn16, S_TOK, LCTX, LCTX, 0);
    // ---- x = x + (ca_attn @ Wo + bo)     (in-place elementwise residual)
    gemm_wmma<EPI_RES_GATE><<<gq, blk, 0, stream>>>(attn16, ca_wo, ca_bo, xout, xout,
                                                    nullptr, nullptr, S_TOK, DIM, DIM, FS);
    // ---- FFN: h = mod(LN(x), e4, e3); x += gate(gelu(h W1 + b1) W2 + b2, e5)
    ln_mod<<<S_TOK, blk, 0, stream>>>(xout, e, modu, 3, 4, y16);
    gemm_wmma<EPI_GELU_F16><<<gf1, blk, 0, stream>>>(y16, ffn_w1, ffn_b1, g16, nullptr, nullptr, nullptr, S_TOK, FFN_D, DIM, FS);
    gemm_wmma<EPI_RES_GATE><<<gq, blk, 0, stream>>>(g16, ffn_w2, ffn_b2, xout, xout,
                                                    modu + 5 * DIM, e + 5 * DIM, S_TOK, DIM, FFN_D, FS);
}
